// Tacotron2_75668733820957
// MI455X (gfx1250) — matrix operand order repaired and checked by reading, not compiled, not run on hardware
//
#include <hip/hip_runtime.h>
#include <hip/hip_bf16.h>
#include <stdint.h>

typedef __attribute__((ext_vector_type(16))) _Float16 v16h;
typedef __attribute__((ext_vector_type(4)))  _Float16 v4h;
typedef __attribute__((ext_vector_type(4)))  float    v4f;
typedef __attribute__((ext_vector_type(8)))  float    v8f;

#define BB    64
#define TENC  192
#define TDEC  600
#define NMELS 80

__device__ __forceinline__ float sigf(float x) { return 1.f / (1.f + expf(-x)); }

// Swizzled LDS position helpers: data is stored exactly where the consuming
// lane's WMMA fragment expects it, so a fragment load is one 32B v16h read.
//  A (16x32 f16): lane = m16 + 16*((k>>3)&1), elem = (k>>4)*8 + (k&7)
//  B (32x16 f16): lane = n16 + 16*((k>>3)&1), elem = (k>>4)*8 + (k&7)
// pos = strip*512 + lane*16 + elem  (strip = 16-row / 16-col group)

// ============================================================================
// Fused (conv-as-)GEMM with WMMA:
//   Y[r,n] = act( (sum_{tap,k} X[b,t+tap-pad,k] * W[n,k,tap] + bias[n]) * scale[n] + offset[n] )
// X row-major [Bn*T, Cin]; W [Cout, Cin, KW]; Y [Bn*T, Cout]. KW=1,pad=0 -> linear.
// Block: 256 threads (8 wave32), 64x64 tile. Requires: Bn*T % 64 == 0, Cin % 4 == 0.
// ============================================================================
__global__ __launch_bounds__(256) void k_conv_gemm(
    const float* __restrict__ X, const float* __restrict__ W,
    const float* __restrict__ bias, const float* __restrict__ scale,
    const float* __restrict__ offset, float* __restrict__ Y,
    int Bn, int T, int Cin, int Cout, int KW, int pad, int act)
{
  __shared__ __align__(32) _Float16 AsSw[2048];
  __shared__ __align__(32) _Float16 BsSw[2048];
  const int tid  = threadIdx.x;
  const int lane = tid & 31;
  const int wave = tid >> 5;
  const int tr   = wave & 3;
  const int tc0  = (wave >> 2) << 1;
  const int m16  = lane & 15;
  const int half = lane >> 4;
  const int nBase = blockIdx.x * 64;
  const int mBase = blockIdx.y * 64;

  v8f acc0 = {}; v8f acc1 = {};

  for (int tap = 0; tap < KW; ++tap) {
    for (int kc = 0; kc < Cin; kc += 32) {
      // ---- stage A chunk (64 rows x 32 k), branchless, b128 loads, swizzled ----
      #pragma unroll
      for (int i = 0; i < 2; ++i) {
        int idx = i * 256 + tid;          // 512 quads
        int row = idx >> 3;               // 0..63
        int c0  = (idx & 7) << 2;         // 0,4,..,28
        int r = mBase + row;
        int b = r / T, t = r - b * T;
        int ts = t + tap - pad;
        int tsc = min(max(ts, 0), T - 1);
        bool rowok = (ts >= 0) && (ts < T);
        int kb = kc + c0;
        bool ok = rowok && (kb < Cin);    // quads never partially valid (Cin%4==0)
        int kbc = (kb < Cin) ? kb : 0;
        const float* src = X + ((size_t)(b * T + tsc)) * Cin + kbc;
        v4f f = *reinterpret_cast<const v4f*>(src);
        v4h h;
        #pragma unroll
        for (int j = 0; j < 4; ++j) h[j] = (_Float16)(ok ? f[j] : 0.f);
        int alane = (row & 15) + (((c0 >> 3) & 1) << 4);
        int aelem = ((c0 >> 4) << 3) + (c0 & 7);
        *reinterpret_cast<v4h*>(&AsSw[((row >> 4) << 9) + (alane << 4) + aelem]) = h;
      }
      // ---- stage B chunk (32 k x 64 n), swizzled ----
      #pragma unroll
      for (int i = 0; i < 2; ++i) {
        int idx = i * 256 + tid;
        int nn  = idx & 63;
        int kq  = (idx >> 6) << 2;        // 0,4,..,28
        int n   = nBase + nn;
        int ncl = min(n, Cout - 1);
        int kb  = kc + kq;
        bool ok = (n < Cout) && (kb < Cin);
        int kbc = (kb < Cin) ? kb : 0;
        v4h h;
        if (KW == 1) {
          v4f f = *reinterpret_cast<const v4f*>(W + (size_t)ncl * Cin + kbc);
          #pragma unroll
          for (int j = 0; j < 4; ++j) h[j] = (_Float16)(ok ? f[j] : 0.f);
        } else {
          #pragma unroll
          for (int j = 0; j < 4; ++j) {
            float v = W[((size_t)ncl * Cin + kbc + j) * KW + tap];
            h[j] = (_Float16)(ok ? v : 0.f);
          }
        }
        int blane = (nn & 15) + (((kq >> 3) & 1) << 4);
        *reinterpret_cast<v4h*>(&BsSw[((nn >> 4) << 9) + (blane << 4) + ((kq >> 4) << 3) + (kq & 7)]) = h;
      }
      __syncthreads();
      // ---- fragments: one 32B LDS read each ----
      v16h a  = *reinterpret_cast<const v16h*>(&AsSw[(tr << 9) + (lane << 4)]);
      v16h bf0 = *reinterpret_cast<const v16h*>(&BsSw[(tc0 << 9) + (lane << 4)]);
      v16h bf1 = *reinterpret_cast<const v16h*>(&BsSw[((tc0 + 1) << 9) + (lane << 4)]);
      acc0 = __builtin_amdgcn_wmma_f32_16x16x32_f16(false, a, false, bf0, (short)0, acc0, false, false);
      acc1 = __builtin_amdgcn_wmma_f32_16x16x32_f16(false, a, false, bf1, (short)0, acc1, false, false);
      __syncthreads();
    }
  }
  // ---- epilogue: bias + BN + activation; C layout: N=lane&15, M=v+8*(lane>>4) ----
  #pragma unroll
  for (int tci = 0; tci < 2; ++tci) {
    v8f acc = tci ? acc1 : acc0;
    int n = nBase + (tc0 + tci) * 16 + m16;
    if (n >= Cout) continue;
    float bv = bias   ? bias[n]   : 0.f;
    float sv = scale  ? scale[n]  : 1.f;
    float ov = offset ? offset[n] : 0.f;
    #pragma unroll
    for (int vv = 0; vv < 8; ++vv) {
      int m = mBase + tr * 16 + vv + (half << 3);
      float y = acc[vv] + bv;
      y = y * sv + ov;
      if (act == 1) y = fmaxf(y, 0.f);
      else if (act == 2) y = tanhf(y);
      Y[(size_t)m * Cout + n] = y;
    }
  }
}

// ============================================================================
// RNN-step GEMM: Z[64,N] = sum_s A_s[64,K_s] @ B_s[:,ko_s:ko_s+K_s]^T + bias
// Up to 3 (A,B) segments. Requires N % 64 == 0, K_s % 32 == 0 here.
// ============================================================================
__global__ __launch_bounds__(256) void k_rnn_gemm(
    const float* A0, int ldA0, int K0, const float* B0, int ldb0, int ko0,
    const float* A1, int ldA1, int K1, const float* B1, int ldb1, int ko1,
    const float* A2, int ldA2, int K2, const float* B2, int ldb2, int ko2,
    const float* __restrict__ bias, float* __restrict__ Z, int N)
{
  __shared__ __align__(32) _Float16 AsSw[2048];
  __shared__ __align__(32) _Float16 BsSw[2048];
  const int tid  = threadIdx.x;
  const int lane = tid & 31;
  const int wave = tid >> 5;
  const int tr   = wave & 3;
  const int tc0  = (wave >> 2) << 1;
  const int m16  = lane & 15;
  const int half = lane >> 4;
  const int nBase = blockIdx.x * 64;

  v8f acc0 = {}; v8f acc1 = {};

  for (int s = 0; s < 3; ++s) {
    const float* A  = (s == 0) ? A0 : (s == 1) ? A1 : A2;
    const float* Bp = (s == 0) ? B0 : (s == 1) ? B1 : B2;
    int ldA = (s == 0) ? ldA0 : (s == 1) ? ldA1 : ldA2;
    int ldb = (s == 0) ? ldb0 : (s == 1) ? ldb1 : ldb2;
    int ko  = (s == 0) ? ko0  : (s == 1) ? ko1  : ko2;
    int Ks  = (s == 0) ? K0   : (s == 1) ? K1   : K2;
    if (Ks <= 0) continue;
    for (int kc = 0; kc < Ks; kc += 32) {
      #pragma unroll
      for (int i = 0; i < 2; ++i) {
        int idx = i * 256 + tid;
        int row = idx >> 3;
        int c0  = (idx & 7) << 2;
        int kb  = kc + c0;                          // always < Ks (Ks % 32 == 0)
        v4f f = *reinterpret_cast<const v4f*>(A + (size_t)row * ldA + kb);
        v4h h;
        #pragma unroll
        for (int j = 0; j < 4; ++j) h[j] = (_Float16)f[j];
        int alane = (row & 15) + (((c0 >> 3) & 1) << 4);
        int aelem = ((c0 >> 4) << 3) + (c0 & 7);
        *reinterpret_cast<v4h*>(&AsSw[((row >> 4) << 9) + (alane << 4) + aelem]) = h;
      }
      #pragma unroll
      for (int i = 0; i < 2; ++i) {
        int idx = i * 256 + tid;
        int nn  = idx & 63;
        int kq  = (idx >> 6) << 2;
        int kb  = kc + kq;
        v4f f = *reinterpret_cast<const v4f*>(Bp + (size_t)(nBase + nn) * ldb + ko + kb);
        v4h h;
        #pragma unroll
        for (int j = 0; j < 4; ++j) h[j] = (_Float16)f[j];
        int blane = (nn & 15) + (((kq >> 3) & 1) << 4);
        *reinterpret_cast<v4h*>(&BsSw[((nn >> 4) << 9) + (blane << 4) + ((kq >> 4) << 3) + (kq & 7)]) = h;
      }
      __syncthreads();
      v16h a  = *reinterpret_cast<const v16h*>(&AsSw[(tr << 9) + (lane << 4)]);
      v16h bf0 = *reinterpret_cast<const v16h*>(&BsSw[(tc0 << 9) + (lane << 4)]);
      v16h bf1 = *reinterpret_cast<const v16h*>(&BsSw[((tc0 + 1) << 9) + (lane << 4)]);
      acc0 = __builtin_amdgcn_wmma_f32_16x16x32_f16(false, a, false, bf0, (short)0, acc0, false, false);
      acc1 = __builtin_amdgcn_wmma_f32_16x16x32_f16(false, a, false, bf1, (short)0, acc1, false, false);
      __syncthreads();
    }
  }
  #pragma unroll
  for (int tci = 0; tci < 2; ++tci) {
    v8f acc = tci ? acc1 : acc0;
    int n = nBase + (tc0 + tci) * 16 + m16;
    float bv = bias ? bias[n] : 0.f;
    #pragma unroll
    for (int vv = 0; vv < 8; ++vv) {
      int m = tr * 16 + vv + (half << 3);
      Z[(size_t)m * N + n] = acc[vv] + bv;
    }
  }
}

// ============================================================================
// Pointwise / small kernels
// ============================================================================
__global__ void k_zero(float* p, int n) {
  int i = blockIdx.x * 256 + threadIdx.x;
  if (i < n) p[i] = 0.f;
}

__global__ void k_embed(const int* __restrict__ texts, const float* __restrict__ emb,
                        float* __restrict__ out) {
  int idx = blockIdx.x * 256 + threadIdx.x;
  if (idx >= BB * TENC * 512) return;
  int r = idx >> 9, c = idx & 511;
  out[idx] = emb[(size_t)texts[r] * 512 + c];
}

__global__ void k_lstm_gate(const float* __restrict__ z, float* __restrict__ H,
                            float* __restrict__ C, float* __restrict__ mem,
                            int t, int dirOff) {
  int idx = blockIdx.x * 256 + threadIdx.x;
  if (idx >= BB * 256) return;
  int b = idx >> 8, j = idx & 255;
  const float* zb = z + (size_t)b * 1024;
  float cn = sigf(zb[256 + j]) * C[idx] + sigf(zb[j]) * tanhf(zb[512 + j]);
  float h = sigf(zb[768 + j]) * tanhf(cn);
  C[idx] = cn; H[idx] = h;
  mem[((size_t)(b * TENC + t)) * 512 + dirOff + j] = h;
}

__global__ void k_dec_gate_q(const float* __restrict__ z, float* __restrict__ H,
                             float* __restrict__ C, const float* __restrict__ att_q,
                             float* __restrict__ Q) {
  __shared__ float sh[512];
  int b = blockIdx.x, j = threadIdx.x;
  const float* zb = z + (size_t)b * 2048;
  float cn = sigf(zb[512 + j]) * C[b * 512 + j] + sigf(zb[j]) * tanhf(zb[1024 + j]);
  float h = sigf(zb[1536 + j]) * tanhf(cn);
  C[b * 512 + j] = cn; H[b * 512 + j] = h; sh[j] = h;
  __syncthreads();
  if (j < 128) {
    const float* w = att_q + (size_t)j * 512;
    float s = 0.f;
    for (int d = 0; d < 512; ++d) s += sh[d] * w[d];
    Q[b * 128 + j] = s;
  }
}

__global__ void k_energy(const float* __restrict__ aw, const float* __restrict__ acum,
                         const float* __restrict__ lw, const float* __restrict__ ll,
                         const float* __restrict__ vvec, const float* __restrict__ Q,
                         const float* __restrict__ pm, const unsigned char* __restrict__ mask,
                         float* __restrict__ E) {
  int idx = blockIdx.x * 256 + threadIdx.x;
  if (idx >= BB * TENC) return;
  int b = idx / TENC, t = idx - b * TENC;
  float loc[32];
  #pragma unroll
  for (int f = 0; f < 32; ++f) loc[f] = 0.f;
  for (int k = 0; k < 31; ++k) {
    int ts = t + k - 15;
    int tsc = min(max(ts, 0), TENC - 1);
    bool ok = (ts >= 0) && (ts < TENC);
    float c0 = ok ? aw[b * TENC + tsc] : 0.f;
    float c1 = ok ? acum[b * TENC + tsc] : 0.f;
    #pragma unroll
    for (int f = 0; f < 32; ++f)
      loc[f] += c0 * lw[f * 62 + k] + c1 * lw[f * 62 + 31 + k];
  }
  const float* qb = Q + b * 128;
  const float* pmb = pm + (size_t)(b * TENC + t) * 128;
  float e = 0.f;
  for (int a = 0; a < 128; ++a) {
    float la = 0.f;
    const float* lla = ll + a * 32;
    #pragma unroll
    for (int f = 0; f < 32; ++f) la += loc[f] * lla[f];
    e += tanhf(qb[a] + pmb[a] + la) * vvec[a];
  }
  if (!mask[idx]) e = -1e9f;
  E[idx] = e;
}

__global__ void k_softmax_ctx(const float* __restrict__ E, const float* __restrict__ mem,
                              float* __restrict__ aw, float* __restrict__ acum,
                              float* __restrict__ ctx, float* __restrict__ aligns, int step) {
  __shared__ float se[TENC];
  __shared__ float red[256];
  int b = blockIdx.x, tid = threadIdx.x;
  float v = (tid < TENC) ? E[b * TENC + tid] : -1e30f;
  red[tid] = v; __syncthreads();
  for (int s = 128; s > 0; s >>= 1) { if (tid < s) red[tid] = fmaxf(red[tid], red[tid + s]); __syncthreads(); }
  float mx = red[0]; __syncthreads();
  float ex = (tid < TENC) ? expf(v - mx) : 0.f;
  red[tid] = ex; __syncthreads();
  for (int s = 128; s > 0; s >>= 1) { if (tid < s) red[tid] += red[tid + s]; __syncthreads(); }
  float inv = 1.f / red[0];
  if (tid < TENC) {
    float al = ex * inv;
    se[tid] = al;
    aw[b * TENC + tid] = al;
    acum[b * TENC + tid] += al;
    aligns[(size_t)(b * TDEC + step) * TENC + tid] = al;
  }
  __syncthreads();
  for (int d = tid; d < 512; d += 256) {
    float s = 0.f;
    for (int t = 0; t < TENC; ++t) s += se[t] * mem[((size_t)(b * TENC + t)) * 512 + d];
    ctx[b * 512 + d] = s;
  }
}

__global__ void k_proj_gate(const float* __restrict__ H, const float* __restrict__ CTX,
                            const float* __restrict__ pw, const float* __restrict__ pb,
                            const float* __restrict__ gw, const float* __restrict__ gb,
                            float* __restrict__ dec, float* __restrict__ gate, int step) {
  int b = blockIdx.x, j = threadIdx.x;
  if (j > 80) return;
  const float* w = (j < 80) ? pw + (size_t)j * 1024 : gw;
  float acc = (j < 80) ? pb[j] : gb[0];
  for (int d = 0; d < 512; ++d) acc += H[b * 512 + d] * w[d];
  for (int d = 0; d < 512; ++d) acc += CTX[b * 512 + d] * w[512 + d];
  if (j < 80) dec[(size_t)(b * TDEC + step) * 80 + j] = acc;
  else        gate[b * TDEC + step] = acc;
}

__global__ void k_build_decin(const float* __restrict__ mels, float* __restrict__ decin) {
  int idx = blockIdx.x * 256 + threadIdx.x;
  if (idx >= TDEC * BB * NMELS) return;
  int r = idx / NMELS, c = idx - r * NMELS;
  int t = r >> 6, b = r & 63;
  decin[idx] = (t == 0) ? 0.f : mels[((size_t)(b * TDEC + t - 1)) * NMELS + c];
}

__global__ void k_residual(const float* __restrict__ y, const float* __restrict__ dec,
                           float* __restrict__ out, int n) {
  int i = blockIdx.x * 256 + threadIdx.x;
  if (i < n) out[i] = y[i] + dec[i];
}

// ============================================================================
// Host orchestration
// ============================================================================
extern "C" void kernel_launch(void* const* d_in, const int* in_sizes, int n_in,
                              void* d_out, int out_size, void* d_ws, size_t ws_size,
                              hipStream_t stream) {
  const int*           texts = (const int*)d_in[0];
  const unsigned char* mask  = (const unsigned char*)d_in[1];
  const float*         mels  = (const float*)d_in[2];
  const float*         emb   = (const float*)d_in[3];
  const float *cw[3], *cb[3], *cs[3], *co[3];
  for (int i = 0; i < 3; ++i) {
    cw[i] = (const float*)d_in[4 + i * 4]; cb[i] = (const float*)d_in[5 + i * 4];
    cs[i] = (const float*)d_in[6 + i * 4]; co[i] = (const float*)d_in[7 + i * 4];
  }
  const float* fWi = (const float*)d_in[16]; const float* fWh = (const float*)d_in[17];
  const float* fb  = (const float*)d_in[18];
  const float* bWi = (const float*)d_in[19]; const float* bWh = (const float*)d_in[20];
  const float* bb  = (const float*)d_in[21];
  const float* pre1 = (const float*)d_in[22]; const float* pre2 = (const float*)d_in[23];
  const float* cWi = (const float*)d_in[24]; const float* cWh = (const float*)d_in[25];
  const float* cbias = (const float*)d_in[26];
  const float* att_q = (const float*)d_in[27]; const float* att_mem = (const float*)d_in[28];
  const float* locw = (const float*)d_in[29]; const float* locl = (const float*)d_in[30];
  const float* attv = (const float*)d_in[31];
  const float* projw = (const float*)d_in[32]; const float* projb = (const float*)d_in[33];
  const float* gatew = (const float*)d_in[34]; const float* gateb = (const float*)d_in[35];
  const float *pw[5], *pb_[5], *ps[5], *po[5];
  for (int i = 0; i < 5; ++i) {
    pw[i] = (const float*)d_in[36 + i * 4]; pb_[i] = (const float*)d_in[37 + i * 4];
    ps[i] = (const float*)d_in[38 + i * 4]; po[i] = (const float*)d_in[39 + i * 4];
  }

  // ---- outputs (concatenated): dec_out, post_out, alignments, gate_out ----
  float* dec_out  = (float*)d_out;                 // 64*600*80
  float* post_out = dec_out + 3072000;             // 64*600*80
  float* aligns   = post_out + 3072000;            // 64*600*192
  float* gates    = aligns + 7372800;              // 64*600

  // ---- workspace carve-up (floats) ----
  float* ws = (float*)d_ws;
  float* XA     = ws;                 // 12288*512
  float* XB     = XA + 6291456;       // 12288*512
  float* MEM    = XB + 6291456;       // 12288*512
  float* PM     = MEM + 6291456;      // 12288*128
  float* DECIN  = PM + 1572864;       // 38400*80
  float* PRETMP = DECIN + 3072000;    // 38400*256
  float* PREALL = PRETMP + 9830400;   // 38400*256
  float* Zb     = PREALL + 9830400;   // 64*2048
  float* Hb     = Zb + 131072;        // 64*512
  float* Cb     = Hb + 32768;         // 64*512
  float* CTX    = Cb + 32768;         // 64*512
  float* AW     = CTX + 32768;        // 64*192
  float* ACUM   = AW + 12288;         // 64*192
  float* Qb     = ACUM + 12288;       // 64*128
  float* Eb     = Qb + 8192;          // 64*192
  // postnet ping/pong reuse encoder + prenet slabs (free by then)
  float* POSTA = XA;     // XA+XB+MEM+PM region >= 38400*512
  float* POSTB = DECIN;  // DECIN+PRETMP+PREALL region >= 38400*512

  // ---- encoder: embedding -> 3x(conv+BN+relu) ----
  k_embed<<<24576, 256, 0, stream>>>(texts, emb, XA);
  k_conv_gemm<<<dim3(8, 192), 256, 0, stream>>>(XA, cw[0], cb[0], cs[0], co[0], XB,
                                                BB, TENC, 512, 512, 5, 2, 1);
  k_conv_gemm<<<dim3(8, 192), 256, 0, stream>>>(XB, cw[1], cb[1], cs[1], co[1], XA,
                                                BB, TENC, 512, 512, 5, 2, 1);
  k_conv_gemm<<<dim3(8, 192), 256, 0, stream>>>(XA, cw[2], cb[2], cs[2], co[2], XB,
                                                BB, TENC, 512, 512, 5, 2, 1);

  // ---- bi-LSTM (sequential scans; graph replay amortizes launches) ----
  k_zero<<<64, 256, 0, stream>>>(Hb, 16384);
  k_zero<<<64, 256, 0, stream>>>(Cb, 16384);
  for (int t = 0; t < TENC; ++t) {
    k_rnn_gemm<<<16, 256, 0, stream>>>(XB + (size_t)t * 512, TENC * 512, 512, fWi, 512, 0,
                                       Hb, 256, 256, fWh, 256, 0,
                                       nullptr, 0, 0, nullptr, 0, 0,
                                       fb, Zb, 1024);
    k_lstm_gate<<<64, 256, 0, stream>>>(Zb, Hb, Cb, MEM, t, 0);
  }
  k_zero<<<64, 256, 0, stream>>>(Hb, 16384);
  k_zero<<<64, 256, 0, stream>>>(Cb, 16384);
  for (int s = 0; s < TENC; ++s) {
    int t = TENC - 1 - s;
    k_rnn_gemm<<<16, 256, 0, stream>>>(XB + (size_t)t * 512, TENC * 512, 512, bWi, 512, 0,
                                       Hb, 256, 256, bWh, 256, 0,
                                       nullptr, 0, 0, nullptr, 0, 0,
                                       bb, Zb, 1024);
    k_lstm_gate<<<64, 256, 0, stream>>>(Zb, Hb, Cb, MEM, t, 256);
  }

  // ---- processed memory: pm = memory @ att_mem^T ----
  k_conv_gemm<<<dim3(2, 192), 256, 0, stream>>>(MEM, att_mem, nullptr, nullptr, nullptr, PM,
                                                12288, 1, 512, 128, 1, 0, 0);

  // ---- prenet for all 600 steps at once (teacher forcing => no recurrence) ----
  k_build_decin<<<12000, 256, 0, stream>>>(mels, DECIN);
  k_conv_gemm<<<dim3(4, 600), 256, 0, stream>>>(DECIN, pre1, nullptr, nullptr, nullptr, PRETMP,
                                                38400, 1, 80, 256, 1, 0, 1);
  k_conv_gemm<<<dim3(4, 600), 256, 0, stream>>>(PRETMP, pre2, nullptr, nullptr, nullptr, PREALL,
                                                38400, 1, 256, 256, 1, 0, 1);

  // ---- decoder scan ----
  k_zero<<<128, 256, 0, stream>>>(Hb, 32768);
  k_zero<<<128, 256, 0, stream>>>(Cb, 32768);
  k_zero<<<128, 256, 0, stream>>>(CTX, 32768);
  k_zero<<<48, 256, 0, stream>>>(AW, 12288);
  k_zero<<<48, 256, 0, stream>>>(ACUM, 12288);
  for (int step = 0; step < TDEC; ++step) {
    k_rnn_gemm<<<32, 256, 0, stream>>>(PREALL + (size_t)step * 64 * 256, 256, 256, cWi, 768, 0,
                                       CTX, 512, 512, cWi, 768, 256,
                                       Hb, 512, 512, cWh, 512, 0,
                                       cbias, Zb, 2048);
    k_dec_gate_q<<<64, 512, 0, stream>>>(Zb, Hb, Cb, att_q, Qb);
    k_energy<<<48, 256, 0, stream>>>(AW, ACUM, locw, locl, attv, Qb, PM, mask, Eb);
    k_softmax_ctx<<<64, 256, 0, stream>>>(Eb, MEM, AW, ACUM, CTX, aligns, step);
    k_proj_gate<<<64, 128, 0, stream>>>(Hb, CTX, projw, projb, gatew, gateb, dec_out, gates, step);
  }

  // ---- postnet: 5x(conv+BN[+tanh]) + residual ----
  k_conv_gemm<<<dim3(8, 600), 256, 0, stream>>>(dec_out, pw[0], pb_[0], ps[0], po[0], POSTA,
                                                BB, TDEC, 80, 512, 5, 2, 2);
  k_conv_gemm<<<dim3(8, 600), 256, 0, stream>>>(POSTA, pw[1], pb_[1], ps[1], po[1], POSTB,
                                                BB, TDEC, 512, 512, 5, 2, 2);
  k_conv_gemm<<<dim3(8, 600), 256, 0, stream>>>(POSTB, pw[2], pb_[2], ps[2], po[2], POSTA,
                                                BB, TDEC, 512, 512, 5, 2, 2);
  k_conv_gemm<<<dim3(8, 600), 256, 0, stream>>>(POSTA, pw[3], pb_[3], ps[3], po[3], POSTB,
                                                BB, TDEC, 512, 512, 5, 2, 2);
  k_conv_gemm<<<dim3(2, 600), 256, 0, stream>>>(POSTB, pw[4], pb_[4], ps[4], po[4], POSTA,
                                                BB, TDEC, 512, 80, 5, 2, 0);
  k_residual<<<12000, 256, 0, stream>>>(POSTA, dec_out, post_out, 3072000);
}
